// NeurJudge_6158983102796
// MI455X (gfx1250) — matrix operand order repaired and checked by reading, not compiled, not run on hardware
//
#include <hip/hip_runtime.h>
#include <hip/hip_bf16.h>
#include <cstdint>
#include <cstddef>

// ---------------------------------------------------------------------------
// NeurJudge on MI455X (gfx1250): bf16 WMMA for every matmul (input projections,
// GRU recurrence, attention score/apply), wave32 kernels, f32 accumulation.
// GRU recurrences run as persistent kernels (one launch per bi-GRU layer) with
// Whh staged in LDS and a device-scope step barrier.
// Workspace requirement ~600 MB (xg arena reused across layers).
// ---------------------------------------------------------------------------

typedef __bf16 v16bf __attribute__((ext_vector_type(16)));
typedef float  v8f   __attribute__((ext_vector_type(8)));

struct __attribute__((aligned(16))) U128 { unsigned int w[4]; };
struct __attribute__((aligned(16))) B256 { U128 lo, hi; };

// A-operand fragment (16x32 bf16): lane<16 holds K {0..7,16..23}, lane>=16 holds
// K {8..15,24..31} of its row (ISA 7.12.2, 16-bit A layout).
__device__ __forceinline__ v16bf frag_a(const __bf16* p) {
  B256 t;
  t.lo = *reinterpret_cast<const U128*>(p);
  t.hi = *reinterpret_cast<const U128*>(p + 16);
  return __builtin_bit_cast(v16bf, t);
}
// B-operand fragment (32x16 bf16): lane<16 holds K {0..7,16..23} of column n,
// lane>=16 holds K {8..15,24..31} (matches A layout).
__device__ __forceinline__ v16bf frag_b(const __bf16* p) {
  B256 t;
  t.lo = *reinterpret_cast<const U128*>(p);
  t.hi = *reinterpret_cast<const U128*>(p + 16);
  return __builtin_bit_cast(v16bf, t);
}
__device__ __forceinline__ v8f wmma_bf16(v16bf a, v16bf b, v8f c) {
  return __builtin_amdgcn_wmma_f32_16x16x32_bf16(false, a, false, b, (short)0, c,
                                                 false, false);
}
__device__ __forceinline__ v8f v8zero() {
  v8f z = {0.f, 0.f, 0.f, 0.f, 0.f, 0.f, 0.f, 0.f};
  return z;
}

// ---------------------------------------------------------------------------
// Generic batched GEMM: C[M,N] = A[M,Kp](bf16) * W[N,Kp]^T(bf16) (+bias), f32 out.
// One wave computes a 32(M)x64(N) tile with 8 WMMA accumulators.
// A rows padded to mult-of-32 with zeros, W rows padded to mult-of-64 with zeros,
// Kp multiple of 32.
// ---------------------------------------------------------------------------
__global__ __launch_bounds__(32) void k_gemm_nt(
    const __bf16* __restrict__ A, const __bf16* __restrict__ W,
    float* __restrict__ C, const float* __restrict__ bias,
    int M, int N, int Kp, int ldc,
    long long sA, long long sW, long long sC) {
  int l = threadIdx.x;
  int n0 = blockIdx.x * 64, m0 = blockIdx.y * 32;
  long long z = blockIdx.z;
  A += z * sA; W += z * sW; C += z * sC;
  int r = l & 15, hi = l >> 4;
  v8f acc[4][2];
#pragma unroll
  for (int i = 0; i < 4; ++i) { acc[i][0] = v8zero(); acc[i][1] = v8zero(); }
  const __bf16* pa0 = A + (long long)(m0 + r) * Kp + hi * 8;
  const __bf16* pa1 = pa0 + (long long)16 * Kp;
  const __bf16* pw  = W + (long long)(n0 + r) * Kp + hi * 8;
  for (int k0 = 0; k0 < Kp; k0 += 32) {
    __builtin_prefetch(pa0 + k0 + 64, 0, 1);  // gfx1250 global_prefetch_b8
    v16bf a0 = frag_a(pa0 + k0);
    v16bf a1 = frag_a(pa1 + k0);
#pragma unroll
    for (int nt = 0; nt < 4; ++nt) {
      v16bf b = frag_b(pw + (long long)(nt * 16) * Kp + k0);
      acc[nt][0] = wmma_bf16(a0, b, acc[nt][0]);
      acc[nt][1] = wmma_bf16(a1, b, acc[nt][1]);
    }
  }
#pragma unroll
  for (int nt = 0; nt < 4; ++nt) {
    int n = n0 + nt * 16 + r;
    if (n >= N) continue;
    float bv = bias ? bias[n] : 0.f;
#pragma unroll
    for (int mt = 0; mt < 2; ++mt) {
#pragma unroll
      for (int g = 0; g < 8; ++g) {
        int m = m0 + mt * 16 + g + hi * 8;
        if (m < M) C[(long long)m * ldc + n] = acc[nt][mt][g] + bv;
      }
    }
  }
}

// ---------------------------------------------------------------------------
// Persistent bidirectional GRU scan. One launch per bi-GRU layer.
//   grid = (ceil(H/16), 2 directions), 1 wave per block.
// Each block owns a 16-wide hidden slice of one direction:
//   - stages its 48 Whh rows (3 gates x 16) into LDS once,
//   - for t in 0..S-1: gates = h @ Whh^T (WMMA from LDS), + xg + bhh,
//     activations, write h_new slice + ys, then device-scope step barrier.
// Whh is ((3H+16) x Hp) bf16 zero-padded; h ping-pong buffers (Nbp x Hp) bf16,
// zero-initialized (pad rows stay zero). counter zero-initialized.
// ---------------------------------------------------------------------------
__global__ __launch_bounds__(32) void k_gru_scan(
    __bf16* __restrict__ hpf, __bf16* __restrict__ hqf,   // fwd ping/pong
    __bf16* __restrict__ hpb, __bf16* __restrict__ hqb,   // bwd ping/pong
    const __bf16* __restrict__ WhhF, const __bf16* __restrict__ WhhB,
    const float* __restrict__ bhhF, const float* __restrict__ bhhB,
    const float* __restrict__ xgF, const float* __restrict__ xgB,
    float* __restrict__ ys, unsigned* __restrict__ counter,
    int Nb, int Nbp, int S, int H, int Hp, int Cout) {
  extern __shared__ __bf16 smem[];  // 48 * Hp bf16
  int l = threadIdx.x;
  int j0 = blockIdx.x * 16;
  int dir = blockIdx.y;  // 0 = forward, 1 = backward
  const __bf16* Whh = dir ? WhhB : WhhF;
  const float* bhh = dir ? bhhB : bhhF;
  const float* xg  = dir ? xgB : xgF;
  __bf16* h0 = dir ? hpb : hpf;
  __bf16* h1 = dir ? hqb : hqf;
  int dirOff = dir ? H : 0;

  // Stage this block's Whh slice into LDS (rows: gate*16 + w  <-  gate*H + j0 + w).
  int total = 48 * Hp;
  for (int e = l; e < total; e += 32) {
    int rrow = e / Hp, col = e - rrow * Hp;
    int gate = rrow >> 4, w = rrow & 15;
    smem[rrow * Hp + col] = Whh[(long long)(gate * H + j0 + w) * Hp + col];
  }
  // single-wave block: LDS producer/consumer is the same wave; compiler waits DScnt.

  int r = l & 15, hi = l >> 4;
  int nblk = gridDim.x * gridDim.y;
  int mtiles = Nbp >> 5;
  int j = j0 + r;

  for (int t = 0; t < S; ++t) {
    const __bf16* h_in = (t & 1) ? h1 : h0;
    __bf16* h_out = (t & 1) ? h0 : h1;
    int s = dir ? (S - 1 - t) : t;
    for (int mb = 0; mb < mtiles; ++mb) {
      int m0 = mb * 32;
      v8f acc[3][2];
#pragma unroll
      for (int g = 0; g < 3; ++g) { acc[g][0] = v8zero(); acc[g][1] = v8zero(); }
      const __bf16* pa0 = h_in + (long long)(m0 + r) * Hp + hi * 8;
      const __bf16* pa1 = pa0 + (long long)16 * Hp;
      for (int k0 = 0; k0 < Hp; k0 += 32) {
        v16bf a0 = frag_a(pa0 + k0);
        v16bf a1 = frag_a(pa1 + k0);
#pragma unroll
        for (int g = 0; g < 3; ++g) {
          v16bf b = frag_b(smem + (g * 16 + r) * Hp + k0 + hi * 8);
          acc[g][0] = wmma_bf16(a0, b, acc[g][0]);
          acc[g][1] = wmma_bf16(a1, b, acc[g][1]);
        }
      }
      if (j < H) {
        float br = bhh[j], bz = bhh[H + j], bn = bhh[2 * H + j];
#pragma unroll
        for (int mt = 0; mt < 2; ++mt) {
#pragma unroll
          for (int g = 0; g < 8; ++g) {
            int b = m0 + mt * 16 + g + hi * 8;
            if (b >= Nb) continue;
            const float* xr = xg + ((long long)b * S + s) * (3 * H);
            float rr = 1.f / (1.f + __expf(-(xr[j]         + acc[0][mt][g] + br)));
            float zz = 1.f / (1.f + __expf(-(xr[H + j]     + acc[1][mt][g] + bz)));
            float nn = tanhf(xr[2 * H + j] + rr * (acc[2][mt][g] + bn));
            float ho = (float)h_in[(long long)b * Hp + j];
            float hn = (1.f - zz) * nn + zz * ho;
            h_out[(long long)b * Hp + j] = (__bf16)hn;
            ys[((long long)b * S + s) * Cout + dirOff + j] = hn;
          }
        }
      }
    }
    // Device-scope generation barrier across all blocks (both directions).
    __threadfence();
    if (l == 0)
      __hip_atomic_fetch_add(counter, 1u, __ATOMIC_RELEASE, __HIP_MEMORY_SCOPE_AGENT);
    unsigned target = (unsigned)(t + 1) * (unsigned)nblk;
    if (l == 0) {
      while (__hip_atomic_load(counter, __ATOMIC_ACQUIRE, __HIP_MEMORY_SCOPE_AGENT) < target)
        __builtin_amdgcn_s_sleep(2);
    }
    __builtin_amdgcn_wave_barrier();
    __threadfence();
  }
}

// ---------------------------------------------------------------------------
// Utility kernels
// ---------------------------------------------------------------------------
__global__ void k_pad_cast(const float* __restrict__ src, __bf16* __restrict__ dst,
                           int rows, int cols, int lds, int pcols,
                           long long sSrc, long long sDst) {
  int rp = blockIdx.x;
  int c = blockIdx.y * 256 + threadIdx.x;
  long long z = blockIdx.z;
  if (c >= pcols) return;
  float v = (rp < rows && c < cols) ? src[z * sSrc + (long long)rp * lds + c] : 0.f;
  dst[z * sDst + (long long)rp * pcols + c] = (__bf16)v;
}

__global__ void k_pad_castT(const float* __restrict__ src, __bf16* __restrict__ dst,
                            int rows, int cols, int lds, int pcols,
                            long long sSrc, long long sDst) {
  int rp = blockIdx.x;
  int c = blockIdx.y * 256 + threadIdx.x;
  long long z = blockIdx.z;
  if (c >= pcols) return;
  float v = (rp < rows && c < cols) ? src[z * sSrc + (long long)c * lds + rp] : 0.f;
  dst[z * sDst + (long long)rp * pcols + c] = (__bf16)v;
}

__global__ void k_gather_pad(const float* __restrict__ emb, const int* __restrict__ idx,
                             __bf16* __restrict__ dst, int M, int Dp, int D) {
  int rp = blockIdx.x;
  int c = blockIdx.y * 256 + threadIdx.x;
  if (c >= Dp) return;
  float v = 0.f;
  if (rp < M && c < D) v = emb[(long long)idx[rp] * D + c];
  dst[(long long)rp * Dp + c] = (__bf16)v;
}

__global__ void k_zero16(__bf16* p, int n) {
  int i = blockIdx.x * 256 + threadIdx.x;
  if (i < n) p[i] = (__bf16)0.f;
}

__global__ void k_zerou32(unsigned* p, int n) {
  int i = blockIdx.x * 256 + threadIdx.x;
  if (i < n) p[i] = 0u;
}

__global__ void k_mean_seq(const float* __restrict__ in, float* __restrict__ out,
                           int Nb, int S, int C) {
  int b = blockIdx.x;
  int c = blockIdx.y * 256 + threadIdx.x;
  if (c >= C) return;
  float s = 0.f;
  for (int t = 0; t < S; ++t) s += in[((long long)b * S + t) * C + c];
  out[(long long)b * C + c] = s / (float)S;
}

__global__ void k_rowmax(const float* __restrict__ in, float* __restrict__ out,
                         int R, int Cc, int ld) {
  int r = blockIdx.x * 256 + threadIdx.x;
  if (r >= R) return;
  const float* p = in + (long long)r * ld;
  float m = -3.4e38f;
  for (int c = 0; c < Cc; ++c) m = fmaxf(m, p[c]);
  out[r] = m;
}

__global__ void k_softmax_s(const float* __restrict__ in, float* __restrict__ out, int S) {
  __shared__ float red[256];
  int b = blockIdx.x, tx = threadIdx.x;
  const float* p = in + (long long)b * S;
  float m = -3.4e38f;
  for (int s = tx; s < S; s += 256) m = fmaxf(m, p[s]);
  red[tx] = m; __syncthreads();
  for (int o = 128; o > 0; o >>= 1) { if (tx < o) red[tx] = fmaxf(red[tx], red[tx + o]); __syncthreads(); }
  m = red[0]; __syncthreads();
  float sum = 0.f;
  for (int s = tx; s < S; s += 256) sum += __expf(p[s] - m);
  red[tx] = sum; __syncthreads();
  for (int o = 128; o > 0; o >>= 1) { if (tx < o) red[tx] += red[tx + o]; __syncthreads(); }
  float inv = 1.f / red[0];
  for (int s = tx; s < S; s += 256) out[(long long)b * S + s] = __expf(p[s] - m) * inv;
}

__global__ void k_wsum(const float* __restrict__ attn, const float* __restrict__ ctx,
                       float* __restrict__ out, int S, int C) {
  int b = blockIdx.x;
  int c = blockIdx.y * 256 + threadIdx.x;
  if (c >= C) return;
  float s = 0.f;
  for (int t = 0; t < S; ++t)
    s += attn[(long long)b * S + t] * ctx[((long long)b * S + t) * C + c];
  out[(long long)b * C + c] = s;
}

// masked softmax over q (C==64), a==0 -> -inf, empty row -> 0; bf16 output.
__global__ __launch_bounds__(32) void k_msoftmax_q(const float* __restrict__ in,
                                                   __bf16* __restrict__ out, int C) {
  long long row = blockIdx.x;
  int l = threadIdx.x;
  const float* p = in + row * C;
  float v0 = p[l], v1 = p[l + 32];
  bool m0 = (v0 == 0.f), m1 = (v1 == 0.f);
  if (m0) v0 = -3.4e38f;
  if (m1) v1 = -3.4e38f;
  float m = fmaxf(v0, v1);
  for (int o = 16; o > 0; o >>= 1) m = fmaxf(m, __shfl_xor(m, o, 32));
  float e0 = m0 ? 0.f : __expf(v0 - m);
  float e1 = m1 ? 0.f : __expf(v1 - m);
  float sum = e0 + e1;
  for (int o = 16; o > 0; o >>= 1) sum += __shfl_xor(sum, o, 32);
  float inv = (sum > 0.f) ? 1.f / sum : 0.f;
  out[row * C + l]      = (__bf16)(e0 * inv);
  out[row * C + l + 32] = (__bf16)(e1 * inv);
}

// fact_sep elementwise: sim = (<c,s>/(<s,s>+1e-10)) * s ; diff = c - sim
__global__ __launch_bounds__(32) void k_fact_sep(const float* __restrict__ circ,
                                                 const float* __restrict__ scen,
                                                 float* __restrict__ sim_out,
                                                 float* __restrict__ diff_out, int C) {
  long long row = blockIdx.x;
  const float* pc = circ + row * C;
  const float* ps = scen + row * C;
  int l = threadIdx.x;
  float x3 = 0.f, x4 = 0.f;
  for (int c = l; c < C; c += 32) { float sv = ps[c]; x3 += pc[c] * sv; x4 += sv * sv; }
  for (int o = 16; o > 0; o >>= 1) { x3 += __shfl_xor(x3, o, 32); x4 += __shfl_xor(x4, o, 32); }
  float coef = x3 / (x4 + 1e-10f);
  for (int c = l; c < C; c += 32) {
    float sv = ps[c] * coef;
    sim_out[row * C + c] = sv;
    diff_out[row * C + c] = pc[c] - sv;
  }
}

__global__ __launch_bounds__(32) void k_graph_layer(const float* __restrict__ lab_in,
                                                    float* __restrict__ lab_out,
                                                    const int* __restrict__ nbr,
                                                    const float* __restrict__ mask,
                                                    int Kn, int C) {
  int i = blockIdx.x;
  int l = threadIdx.x;
  const float* Li = lab_in + (long long)i * C;
  float accv[10];
#pragma unroll
  for (int k = 0; k < 10; ++k) accv[k] = 0.f;
  float cnt = 0.f;
  for (int j = 0; j < Kn; ++j) {
    float mj = mask[(long long)i * Kn + j];
    cnt += mj;
    const float* Lj = lab_in + (long long)nbr[(long long)i * Kn + j] * C;
    float x1 = 0.f, x2 = 0.f;
    for (int c = l; c < C; c += 32) { float a = Li[c], bb = Lj[c]; x1 += a * bb; x2 += bb * bb; }
    for (int o = 16; o > 0; o >>= 1) { x1 += __shfl_xor(x1, o, 32); x2 += __shfl_xor(x2, o, 32); }
    float coef = x1 / (x2 + 1e-10f) * mj;
    int k = 0;
    for (int c = l; c < C; c += 32, ++k) accv[k] += coef * Lj[c];
  }
  float denom = fmaxf(cnt, 1.f);
  int k = 0;
  for (int c = l; c < C; c += 32, ++k) {
    float mean = accv[k] / denom;
    lab_out[(long long)i * C + c] = (cnt > 0.f) ? (Li[c] - mean) : Li[c];
  }
}

// concat up to 4 f32 sources into padded bf16 matrix; bX=1 -> broadcast over s.
__global__ void k_cat4(__bf16* __restrict__ dst, int pcols, int S,
                       const float* s0, int w0, int b0,
                       const float* s1, int w1, int b1,
                       const float* s2, int w2, int b2,
                       const float* s3, int w3, int b3) {
  long long r = blockIdx.x;
  int c = blockIdx.y * 256 + threadIdx.x;
  if (c >= pcols) return;
  long long b = r / S;
  float v = 0.f;
  int cc = c;
  if (cc < w0) v = s0[(b0 ? b * w0 : r * (long long)w0) + cc];
  else { cc -= w0;
    if (cc < w1) v = s1[(b1 ? b * w1 : r * (long long)w1) + cc];
    else { cc -= w1;
      if (cc < w2) v = s2[(b2 ? b * w2 : r * (long long)w2) + cc];
      else { cc -= w2;
        if (w3 > 0 && cc < w3) v = s3[(b3 ? b * w3 : r * (long long)w3) + cc];
      }
    }
  }
  dst[r * pcols + c] = (__bf16)v;
}

// out[r,n] = bias[n] + dot([x0|x1|x2], W[n,:])  (f32, small heads)
__global__ void k_linear3(const float* x0, int w0, const float* x1, int w1,
                          const float* x2, int w2, const float* __restrict__ W,
                          const float* __restrict__ bias, float* __restrict__ out,
                          int R, int N) {
  int id = blockIdx.x * 256 + threadIdx.x;
  if (id >= R * N) return;
  int rr = id / N, n = id % N;
  int K = w0 + w1 + w2;
  const float* wr = W + (long long)n * K;
  float s = bias[n];
  for (int k = 0; k < w0; ++k) s += x0[(long long)rr * w0 + k] * wr[k];
  for (int k = 0; k < w1; ++k) s += x1[(long long)rr * w1 + k] * wr[w0 + k];
  for (int k = 0; k < w2; ++k) s += x2[(long long)rr * w2 + k] * wr[w0 + w1 + k];
  out[(long long)rr * N + n] = s;
}

// ---------------------------------------------------------------------------
// Host side
// ---------------------------------------------------------------------------
namespace {

struct Gru8 {
  const float *WihF, *WhhF, *bihF, *bhhF, *WihB, *WhhB, *bihB, *bhhB;
};

struct GruPrepped {
  const __bf16 *WihF, *WihB, *WhhF, *WhhB;
  int din, dinp, h, hp, N, Nw;
};

inline int ceil_to(int v, int m) { return (v + m - 1) / m * m; }

Gru8 grab_gru(void* const* d_in, const int* in_sizes, int& i, int din, int h) {
  Gru8 g;
  if (in_sizes[i] == 3 * h * din) {  // params passed as 8 separate arrays
    g.WihF = (const float*)d_in[i++]; g.WhhF = (const float*)d_in[i++];
    g.bihF = (const float*)d_in[i++]; g.bhhF = (const float*)d_in[i++];
    g.WihB = (const float*)d_in[i++]; g.WhhB = (const float*)d_in[i++];
    g.bihB = (const float*)d_in[i++]; g.bhhB = (const float*)d_in[i++];
  } else {  // one flat buffer in p0..p7 order
    const float* base = (const float*)d_in[i++];
    g.WihF = base; base += (size_t)3 * h * din;
    g.WhhF = base; base += (size_t)3 * h * h;
    g.bihF = base; base += (size_t)3 * h;
    g.bhhF = base; base += (size_t)3 * h;
    g.WihB = base; base += (size_t)3 * h * din;
    g.WhhB = base; base += (size_t)3 * h * h;
    g.bihB = base; base += (size_t)3 * h;
    g.bhhB = base;
  }
  return g;
}

void prep_gru(hipStream_t st, const Gru8& g, GruPrepped& P,
              __bf16* wif, __bf16* wib, __bf16* whf, __bf16* whb) {
  dim3 g1(P.Nw, (P.dinp + 255) / 256, 1);
  k_pad_cast<<<g1, 256, 0, st>>>(g.WihF, wif, P.N, P.din, P.din, P.dinp, 0, 0);
  k_pad_cast<<<g1, 256, 0, st>>>(g.WihB, wib, P.N, P.din, P.din, P.dinp, 0, 0);
  dim3 g2(3 * P.h + 16, (P.hp + 255) / 256, 1);
  k_pad_cast<<<g2, 256, 0, st>>>(g.WhhF, whf, 3 * P.h, P.h, P.h, P.hp, 0, 0);
  k_pad_cast<<<g2, 256, 0, st>>>(g.WhhB, whb, 3 * P.h, P.h, P.h, P.hp, 0, 0);
  P.WihF = wif; P.WihB = wib; P.WhhF = whf; P.WhhB = whb;
}

void run_bigru(hipStream_t st, const GruPrepped& P, const Gru8& g,
               const __bf16* Aact, int Nb, int S,
               float* xgF, float* xgB, __bf16* hbuf, unsigned* counter, float* ys) {
  int M = Nb * S;
  dim3 gg((P.N + 63) / 64, (M + 31) / 32, 1);
  k_gemm_nt<<<gg, 32, 0, st>>>(Aact, P.WihF, xgF, g.bihF, M, P.N, P.dinp, P.N, 0, 0, 0);
  k_gemm_nt<<<gg, 32, 0, st>>>(Aact, P.WihB, xgB, g.bihB, M, P.N, P.dinp, P.N, 0, 0, 0);
  int Nbp = ceil_to(Nb, 32);
  int hn = Nbp * P.hp;
  __bf16* hpf = hbuf;          __bf16* hqf = hbuf + hn;
  __bf16* hpb = hbuf + 2 * hn; __bf16* hqb = hbuf + 3 * hn;
  k_zero16<<<(4 * hn + 255) / 256, 256, 0, st>>>(hbuf, 4 * hn);
  k_zerou32<<<1, 256, 0, st>>>(counter, 1);
  dim3 gs((P.h + 15) / 16, 2, 1);
  size_t shmem = (size_t)48 * P.hp * sizeof(__bf16);
  k_gru_scan<<<gs, 32, shmem, st>>>(hpf, hqf, hpb, hqb, P.WhhF, P.WhhB,
                                    g.bhhF, g.bhhB, xgF, xgB, ys, counter,
                                    Nb, Nbp, S, P.h, P.hp, 2 * P.h);
}

}  // namespace

extern "C" void kernel_launch(void* const* d_in, const int* in_sizes, int n_in,
                              void* d_out, int out_size, void* d_ws, size_t ws_size,
                              hipStream_t stream) {
  (void)n_in; (void)out_size; (void)ws_size;
  const hipStream_t st = stream;

  // ---- dims -------------------------------------------------------------
  const int B = 32, Sd = 512, Ll = 50, Lv = 64, KN = 8;
  const int D = 200, H = 150, NC = 115, NA = 99;
  const int H2 = 300, Hc = 450, Ha = 600;
  const int Dp = 224, Hp = 160, Hcp = 480, Hap = 608;
  const int K2H = 320, K6 = 928, K8 = 1216;
  const int Mdoc = B * Sd;                 // 16384 (mult of 32)
  const int MchV = NC * Ll, Mch = ceil_to(MchV, 32);   // 5750 -> 5760
  const int MarV = NA * Ll, Mar = ceil_to(MarV, 32);   // 4950 -> 4960
  const int Mv = B * Lv;                   // 2048

  // ---- inputs -----------------------------------------------------------
  int i = 0;
  const float* embs = (const float*)d_in[i++];
  Gru8 enc  = grab_gru(d_in, in_sizes, i, D, H);
  Gru8 encc = grab_gru(d_in, in_sizes, i, D, H);
  Gru8 enct = grab_gru(d_in, in_sizes, i, 6 * H, 3 * H);
  Gru8 enca = grab_gru(d_in, in_sizes, i, 8 * H, 4 * H);
  const float* Wc = (const float*)d_in[i++];
  const float* bc = (const float*)d_in[i++];
  const float* Wa = (const float*)d_in[i++];
  const float* ba = (const float*)d_in[i++];
  const float* Wt = (const float*)d_in[i++];
  const float* bt = (const float*)d_in[i++];
  const int* charge   = (const int*)d_in[i++];
  const int* article  = (const int*)d_in[i++];
  const int* documents = (const int*)d_in[i++];
  const int* verd_c = (const int*)d_in[i++];
  const int* verd_a = (const int*)d_in[i++];
  const int* ch_nbr = (const int*)d_in[i++];
  const float* ch_mask = (const float*)d_in[i++];
  const int* ar_nbr = (const int*)d_in[i++];
  const float* ar_mask = (const float*)d_in[i++];
  float* out = (float*)d_out;

  // ---- workspace arena --------------------------------------------------
  char* base = (char*)d_ws;
  size_t off = 0;
  auto alloc = [&](size_t bytes) -> void* {
    void* p = base + off;
    off = (off + bytes + 255) & ~(size_t)255;
    return p;
  };
  auto a16 = [&](size_t elems) -> __bf16* { return (__bf16*)alloc(elems * 2); };
  auto a32 = [&](size_t elems) -> float*  { return (float*)alloc(elems * 4); };

  // weight buffers (bf16, padded)
  GruPrepped Penc  = {nullptr, nullptr, nullptr, nullptr, D, Dp, H, Hp, 3 * H, 512};
  GruPrepped Pencc = Penc;
  GruPrepped Penct = {nullptr, nullptr, nullptr, nullptr, 6 * H, K6, Hc, Hcp, 3 * Hc, 1408};
  GruPrepped Penca = {nullptr, nullptr, nullptr, nullptr, 8 * H, K8, Ha, Hap, 3 * Ha, 1856};
  __bf16* w_enc[4]  = {a16(512 * 224), a16(512 * 224), a16(466 * 160), a16(466 * 160)};
  __bf16* w_encc[4] = {a16(512 * 224), a16(512 * 224), a16(466 * 160), a16(466 * 160)};
  __bf16* w_enct[4] = {a16((size_t)1408 * 928), a16((size_t)1408 * 928),
                       a16((size_t)1366 * 480), a16((size_t)1366 * 480)};
  __bf16* w_enca[4] = {a16((size_t)1856 * 1216), a16((size_t)1856 * 1216),
                       a16((size_t)1816 * 608), a16((size_t)1816 * 608)};

  __bf16* Adoc = a16((size_t)Mdoc * Dp);
  __bf16* Ach  = a16((size_t)Mch * Dp);
  __bf16* Aar  = a16((size_t)Mar * Dp);
  __bf16* Avc  = a16((size_t)Mv * Dp);
  __bf16* Ava  = a16((size_t)Mv * Dp);

  float* xgF = a32((size_t)Mdoc * 1800);  // shared x-gate arena (max layer)
  float* xgB = a32((size_t)Mdoc * 1800);

  float* dh     = a32((size_t)Mdoc * H2);
  float* ch_ys  = a32((size_t)MchV * H2);
  float* ar_ys  = a32((size_t)MarV * H2);
  float* vh     = a32((size_t)Mv * H2);
  float* vh2    = a32((size_t)Mv * H2);
  float* ysbig  = a32((size_t)Mdoc * 1200);  // fla then term (reused)

  __bf16* hbuf = a16((size_t)4 * 128 * Hap);
  unsigned* counter = (unsigned*)alloc(256);

  float* chg   = a32((size_t)NC * H2);   // _charge
  float* artl  = a32((size_t)NA * H2);   // _article
  float* gtmp  = a32((size_t)NC * H2);
  float* nchg  = a32((size_t)NC * H2);   // new_charge
  float* nartl = a32((size_t)NA * H2);   // new_article
  float* mean_d   = a32((size_t)B * H2);
  float* cw1 = a32((size_t)B * H2);
  float* cw2 = a32((size_t)B * H2);
  float* cw3 = a32((size_t)B * H2);
  float* cw4 = a32((size_t)B * H2);

  __bf16* dhbf  = a16((size_t)Mdoc * K2H);
  __bf16* secbf = a16((size_t)Mdoc * K2H);
  __bf16* labbf = a16((size_t)128 * K2H);
  float* Smat = a32((size_t)Mdoc * NC);
  float* rmax = a32((size_t)Mdoc);
  float* attn = a32((size_t)B * Sd);

  __bf16* vhbf  = a16((size_t)B * 64 * K2H);
  __bf16* vhTbf = a16((size_t)B * 320 * 64);
  float* amat   = a32((size_t)Mdoc * 64);
  __bf16* ambf  = a16((size_t)Mdoc * 64);
  float* scen = a32((size_t)Mdoc * H2);
  float* adc  = a32((size_t)Mdoc * H2);
  float* sec  = a32((size_t)Mdoc * H2);
  float* ssc  = a32((size_t)Mdoc * H2);
  float* dsc  = a32((size_t)Mdoc * H2);

  __bf16* catbuf = a16((size_t)Mdoc * K8);  // fact_article, then term input
  float* mean_fla = a32((size_t)B * 1200);
  float* mean_tm  = a32((size_t)B * 900);

  // ---- 1. weight prep ----------------------------------------------------
  prep_gru(st, enc,  Penc,  w_enc[0],  w_enc[1],  w_enc[2],  w_enc[3]);
  prep_gru(st, encc, Pencc, w_encc[0], w_encc[1], w_encc[2], w_encc[3]);
  prep_gru(st, enct, Penct, w_enct[0], w_enct[1], w_enct[2], w_enct[3]);
  prep_gru(st, enca, Penca, w_enca[0], w_enca[1], w_enca[2], w_enca[3]);

  // ---- 2. embedding gathers (f32 -> padded bf16) -------------------------
  dim3 gth(Mdoc, (Dp + 255) / 256, 1);
  k_gather_pad<<<gth, 256, 0, st>>>(embs, documents, Adoc, Mdoc, Dp, D);
  dim3 gch(Mch, (Dp + 255) / 256, 1);
  k_gather_pad<<<gch, 256, 0, st>>>(embs, charge, Ach, MchV, Dp, D);
  dim3 gar(Mar, (Dp + 255) / 256, 1);
  k_gather_pad<<<gar, 256, 0, st>>>(embs, article, Aar, MarV, Dp, D);
  dim3 gv(Mv, (Dp + 255) / 256, 1);
  k_gather_pad<<<gv, 256, 0, st>>>(embs, verd_c, Avc, Mv, Dp, D);
  k_gather_pad<<<gv, 256, 0, st>>>(embs, verd_a, Ava, Mv, Dp, D);

  // ---- 3. label encoders + means -----------------------------------------
  run_bigru(st, Pencc, encc, Ach, NC, Ll, xgF, xgB, hbuf, counter, ch_ys);
  k_mean_seq<<<dim3(NC, (H2 + 255) / 256, 1), 256, 0, st>>>(ch_ys, chg, NC, Ll, H2);
  run_bigru(st, Pencc, encc, Aar, NA, Ll, xgF, xgB, hbuf, counter, ar_ys);
  k_mean_seq<<<dim3(NA, (H2 + 255) / 256, 1), 256, 0, st>>>(ar_ys, artl, NA, Ll, H2);

  // ---- 4. graph decomposition (2 layers each) ----------------------------
  k_graph_layer<<<NC, 32, 0, st>>>(chg, gtmp, ch_nbr, ch_mask, KN, H2);
  k_graph_layer<<<NC, 32, 0, st>>>(gtmp, nchg, ch_nbr, ch_mask, KN, H2);
  k_graph_layer<<<NA, 32, 0, st>>>(artl, gtmp, ar_nbr, ar_mask, KN, H2);
  k_graph_layer<<<NA, 32, 0, st>>>(gtmp, nartl, ar_nbr, ar_mask, KN, H2);

  // ---- 5. document encoder -----------------------------------------------
  run_bigru(st, Penc, enc, Adoc, B, Sd, xgF, xgB, hbuf, counter, dh);
  k_mean_seq<<<dim3(B, (H2 + 255) / 256, 1), 256, 0, st>>>(dh, mean_d, B, Sd, H2);
  k_pad_cast<<<dim3(Mdoc, (K2H + 255) / 256, 1), 256, 0, st>>>(dh, dhbf, Mdoc, H2, H2, K2H, 0, 0);

  // ---- 6. code_wise attention x4 -----------------------------------------
  const float* labs[4] = {nchg, chg, nartl, artl};
  const int labn[4] = {NC, NC, NA, NA};
  float* cws[4] = {cw1, cw2, cw3, cw4};
  for (int q = 0; q < 4; ++q) {
    k_pad_cast<<<dim3(128, (K2H + 255) / 256, 1), 256, 0, st>>>(labs[q], labbf, labn[q], H2, H2, K2H, 0, 0);
    dim3 gg((labn[q] + 63) / 64, Mdoc / 32, 1);
    k_gemm_nt<<<gg, 32, 0, st>>>(dhbf, labbf, Smat, nullptr, Mdoc, labn[q], K2H, labn[q], 0, 0, 0);
    k_rowmax<<<(Mdoc + 255) / 256, 256, 0, st>>>(Smat, rmax, Mdoc, labn[q], labn[q]);
    k_softmax_s<<<B, 256, 0, st>>>(rmax, attn, Sd);
    k_wsum<<<dim3(B, (H2 + 255) / 256, 1), 256, 0, st>>>(attn, dh, cws[q], Sd, H2);
  }

  // ---- 7. fact_sep #1: vh = bigru(verd_c, enc); mask_attn(vh, d_hidden) --
  run_bigru(st, Penc, enc, Avc, B, Lv, xgF, xgB, hbuf, counter, vh);
  {
    k_pad_cast<<<dim3(64, (K2H + 255) / 256, B), 256, 0, st>>>(vh, vhbf, 64, H2, H2, K2H,
                                                              (long long)64 * H2, (long long)64 * K2H);
    dim3 ga(1, Sd / 32, B);
    k_gemm_nt<<<ga, 32, 0, st>>>(dhbf, vhbf, amat, nullptr, Sd, 64, K2H, 64,
                                 (long long)Sd * K2H, (long long)64 * K2H, (long long)Sd * 64);
    k_msoftmax_q<<<Mdoc, 32, 0, st>>>(amat, ambf, 64);
    k_pad_castT<<<dim3(320, 1, B), 256, 0, st>>>(vh, vhTbf, H2, 64, H2, 64,
                                                 (long long)64 * H2, (long long)320 * 64);
    dim3 gs2((H2 + 63) / 64, Sd / 32, B);
    k_gemm_nt<<<gs2, 32, 0, st>>>(ambf, vhTbf, scen, nullptr, Sd, H2, 64, H2,
                                  (long long)Sd * 64, (long long)320 * 64, (long long)Sd * H2);
    k_fact_sep<<<Mdoc, 32, 0, st>>>(dh, scen, adc, sec, H2);
  }

  // ---- 8. charge head ----------------------------------------------------
  k_linear3<<<(B * NC + 255) / 256, 256, 0, st>>>(mean_d, H2, cw1, H2, cw2, H2, Wc, bc, out, B, NC);

  // ---- 9. fact_article -> enca bigru -> article head ---------------------
  k_cat4<<<dim3(Mdoc, (K8 + 255) / 256, 1), 256, 0, st>>>(catbuf, K8, Sd,
      dh, H2, 0, cw3, H2, 1, adc, H2, 0, cw4, H2, 1);
  run_bigru(st, Penca, enca, catbuf, B, Sd, xgF, xgB, hbuf, counter, ysbig);
  k_mean_seq<<<dim3(B, (1200 + 255) / 256, 1), 256, 0, st>>>(ysbig, mean_fla, B, Sd, 1200);
  k_linear3<<<(B * NA + 255) / 256, 256, 0, st>>>(mean_fla, 1200, nullptr, 0, nullptr, 0,
                                                  Wa, ba, out + B * NC, B, NA);

  // ---- 10. fact_sep #2: vh2 = bigru(verd_a, enc); mask_attn(vh2, sec) ----
  run_bigru(st, Penc, enc, Ava, B, Lv, xgF, xgB, hbuf, counter, vh2);
  k_pad_cast<<<dim3(Mdoc, (K2H + 255) / 256, 1), 256, 0, st>>>(sec, secbf, Mdoc, H2, H2, K2H, 0, 0);
  {
    k_pad_cast<<<dim3(64, (K2H + 255) / 256, B), 256, 0, st>>>(vh2, vhbf, 64, H2, H2, K2H,
                                                               (long long)64 * H2, (long long)64 * K2H);
    dim3 ga(1, Sd / 32, B);
    k_gemm_nt<<<ga, 32, 0, st>>>(secbf, vhbf, amat, nullptr, Sd, 64, K2H, 64,
                                 (long long)Sd * K2H, (long long)64 * K2H, (long long)Sd * 64);
    k_msoftmax_q<<<Mdoc, 32, 0, st>>>(amat, ambf, 64);
    k_pad_castT<<<dim3(320, 1, B), 256, 0, st>>>(vh2, vhTbf, H2, 64, H2, 64,
                                                 (long long)64 * H2, (long long)320 * 64);
    dim3 gs2((H2 + 63) / 64, Sd / 32, B);
    k_gemm_nt<<<gs2, 32, 0, st>>>(ambf, vhTbf, scen, nullptr, Sd, H2, 64, H2,
                                  (long long)Sd * 64, (long long)320 * 64, (long long)Sd * H2);
    k_fact_sep<<<Mdoc, 32, 0, st>>>(sec, scen, ssc, dsc, H2);
  }

  // ---- 11. term branch: enct bigru over [dh, ssc, dsc] -> time head ------
  k_cat4<<<dim3(Mdoc, (K6 + 255) / 256, 1), 256, 0, st>>>(catbuf, K6, Sd,
      dh, H2, 0, ssc, H2, 0, dsc, H2, 0, nullptr, 0, 0);
  run_bigru(st, Penct, enct, catbuf, B, Sd, xgF, xgB, hbuf, counter, ysbig);
  k_mean_seq<<<dim3(B, (900 + 255) / 256, 1), 256, 0, st>>>(ysbig, mean_tm, B, Sd, 900);
  k_linear3<<<(B * 11 + 255) / 256, 256, 0, st>>>(mean_tm, 900, nullptr, 0, nullptr, 0,
                                                  Wt, bt, out + B * NC + B * NA, B, 11);
}
